// AssignAttention_19670950216054
// MI455X (gfx1250) — matrix operand order repaired and checked by reading, not compiled, not run on hardware
//
#include <hip/hip_runtime.h>
#include <hip/hip_bf16.h>
#include <cstdint>
#include <cstddef>

// Problem constants (from reference)
#define B_   16
#define N_   64
#define S_   4096
#define C_   768
#define H_   8
#define HD_  96

typedef __bf16 bf16_t;
typedef __bf16 v16bf __attribute__((ext_vector_type(16)));
typedef float  v8f   __attribute__((ext_vector_type(8)));
typedef int    v4i_g __attribute__((vector_size(16)));   // matches builtin's V4i

union Frag32B {
    struct { uint4 lo, hi; } q;
    v16bf v;
};

// Load 16 bf16 (one WMMA operand's per-lane data) as two 16B chunks.
__device__ __forceinline__ v16bf load_bf16_frag(const bf16_t* p0, const bf16_t* p1) {
    Frag32B f;
    f.q.lo = *(const uint4*)p0;
    f.q.hi = *(const uint4*)p1;
    return f.v;
}

// A-fragment (16x32 bf16) from fp32 rows with on-the-fly cvt.
// Per-lane chunks: c0..c0+7 and c0+16..c0+23 (ISA 16-bit A layout).
__device__ __forceinline__ v16bf make_a_from_f32(const float* __restrict__ row, int c0) {
    const float4* p0 = (const float4*)(row + c0);
    const float4* p1 = (const float4*)(row + c0 + 16);
    float4 x0 = p0[0], x1 = p0[1];
    float4 y0 = p1[0], y1 = p1[1];
    v16bf a;
    a[0]  = (bf16_t)x0.x; a[1]  = (bf16_t)x0.y; a[2]  = (bf16_t)x0.z; a[3]  = (bf16_t)x0.w;
    a[4]  = (bf16_t)x1.x; a[5]  = (bf16_t)x1.y; a[6]  = (bf16_t)x1.z; a[7]  = (bf16_t)x1.w;
    a[8]  = (bf16_t)y0.x; a[9]  = (bf16_t)y0.y; a[10] = (bf16_t)y0.z; a[11] = (bf16_t)y0.w;
    a[12] = (bf16_t)y1.x; a[13] = (bf16_t)y1.y; a[14] = (bf16_t)y1.z; a[15] = (bf16_t)y1.w;
    return a;
}

// Same, but scale the fp32 values before conversion (used to fold 1/(count+1)).
__device__ __forceinline__ v16bf make_a_scaled(const float* __restrict__ row, int c0, float s) {
    const float4* p0 = (const float4*)(row + c0);
    const float4* p1 = (const float4*)(row + c0 + 16);
    float4 x0 = p0[0], x1 = p0[1];
    float4 y0 = p1[0], y1 = p1[1];
    v16bf a;
    a[0]  = (bf16_t)(x0.x*s); a[1]  = (bf16_t)(x0.y*s); a[2]  = (bf16_t)(x0.z*s); a[3]  = (bf16_t)(x0.w*s);
    a[4]  = (bf16_t)(x1.x*s); a[5]  = (bf16_t)(x1.y*s); a[6]  = (bf16_t)(x1.z*s); a[7]  = (bf16_t)(x1.w*s);
    a[8]  = (bf16_t)(y0.x*s); a[9]  = (bf16_t)(y0.y*s); a[10] = (bf16_t)(y0.z*s); a[11] = (bf16_t)(y0.w*s);
    a[12] = (bf16_t)(y1.x*s); a[13] = (bf16_t)(y1.y*s); a[14] = (bf16_t)(y1.z*s); a[15] = (bf16_t)(y1.w*s);
    return a;
}

__device__ __forceinline__ v8f wmma_bf16(v16bf a, v16bf b, v8f c) {
    return __builtin_amdgcn_wmma_f32_16x16x32_bf16(
        /*neg_a=*/false, a, /*neg_b=*/false, b,
        /*c_mod=*/(short)0, c, /*reuse_a=*/false, /*reuse_b=*/false);
}

// ---------------------------------------------------------------------------
// Async global -> LDS copy (CDNA5 ASYNCcnt path), with safe fallback.
// Builtin signature (from diagnostics): (v4i AS1*, v4i AS3*, imm int, imm int).
// ---------------------------------------------------------------------------
#if defined(__HIP_DEVICE_COMPILE__) && __has_builtin(__builtin_amdgcn_global_load_async_to_lds_b128)
#define ASYNC_LDS 1
#else
#define ASYNC_LDS 0
#endif

__device__ __forceinline__ void async_copy16(bf16_t* ldst, const bf16_t* gsrc) {
#if ASYNC_LDS
    __builtin_amdgcn_global_load_async_to_lds_b128(
        (__attribute__((address_space(1))) v4i_g*)gsrc,
        (__attribute__((address_space(3))) v4i_g*)ldst, 0, 0);
#else
    *(uint4*)ldst = *(const uint4*)gsrc;
#endif
}

__device__ __forceinline__ void wait_async0() {
#if ASYNC_LDS
    asm volatile("s_wait_asynccnt 0" ::: "memory");
#else
    asm volatile("" ::: "memory");
#endif
}

// ---------------------------------------------------------------------------
// Prep kernels
// ---------------------------------------------------------------------------
__global__ void prep_zero(float* __restrict__ p, int n) {
    int i = blockIdx.x * 256 + threadIdx.x;
    if (i < n) p[i] = 0.f;
}

__global__ void prep_cvt4(const float* __restrict__ wq, const float* __restrict__ wk,
                          const float* __restrict__ wv, const float* __restrict__ wo,
                          bf16_t* __restrict__ q, bf16_t* __restrict__ k,
                          bf16_t* __restrict__ v, bf16_t* __restrict__ o, int n) {
    int i = blockIdx.x * 256 + threadIdx.x;
    if (i < n) {
        q[i] = (bf16_t)wq[i];
        k[i] = (bf16_t)wk[i];
        v[i] = (bf16_t)wv[i];
        o[i] = (bf16_t)wo[i];
    }
}

// q projection on WMMA: [64 n x 768 c] x [768 c x 768 d] per batch.
// Grid (2, B): blockIdx.x picks half of the d range (24 tiles; wave -> 3 tiles).
__global__ __launch_bounds__(256) void qproj_wmma(
    const float* __restrict__ query, const bf16_t* __restrict__ WqB,
    bf16_t* __restrict__ qB)
{
    const int tid  = threadIdx.x;
    const int lane = tid & 31;
    const int wave = tid >> 5;
    const int col  = lane & 15;
    const int hi16 = lane >> 4;
    const int b    = blockIdx.y;

    v8f acc[4][3];
    for (int mt = 0; mt < 4; ++mt)
        for (int j = 0; j < 3; ++j)
            for (int r = 0; r < 8; ++r) acc[mt][j][r] = 0.f;

    for (int kk = 0; kk < 24; ++kk) {
        v16bf a[4];
        for (int mt = 0; mt < 4; ++mt)
            a[mt] = make_a_from_f32(query + (size_t)(b * N_ + mt * 16 + col) * C_,
                                    kk * 32 + hi16 * 8);
        for (int j = 0; j < 3; ++j) {
            int dg0 = (blockIdx.x * 24 + wave * 3 + j) * 16;
            const bf16_t* pb = WqB + (size_t)(dg0 + col) * C_ + kk * 32 + hi16 * 8;
            v16bf bf = load_bf16_frag(pb, pb + 16);
            for (int mt = 0; mt < 4; ++mt)
                acc[mt][j] = wmma_bf16(a[mt], bf, acc[mt][j]);
        }
    }
    for (int j = 0; j < 3; ++j) {
        int dg = (blockIdx.x * 24 + wave * 3 + j) * 16 + col;
        int h = dg / HD_, d = dg % HD_;
        for (int mt = 0; mt < 4; ++mt)
            for (int r = 0; r < 8; ++r) {
                int n = mt * 16 + hi16 * 8 + r;
                qB[((size_t)(b * H_ + h) * N_ + n) * HD_ + d] = (bf16_t)acc[mt][j][r];
            }
    }
}

// ---------------------------------------------------------------------------
// Main fused kernel: async-staged weights (LDS, double buffered) -> K/V WMMA
// projection -> logits WMMA -> wave32 argmax -> LDS scatter -> global accum.
// Grid: (S/128, B), 256 threads = 8 waves, each wave owns 16 tokens.
// ---------------------------------------------------------------------------
__global__ __launch_bounds__(256) void assign_attn_main(
    const float*  __restrict__ key_in,  // [B,S,C] fp32
    const bf16_t* __restrict__ WkB,     // [C,C] bf16
    const bf16_t* __restrict__ WvB,     // [C,C] bf16
    const bf16_t* __restrict__ qB,      // [B,H,N,HD] bf16
    float* __restrict__ accumG,         // [B,H,N,HD]
    float* __restrict__ countG)         // [B,H,N]
{
    // Phase-shared LDS region (55296 B):
    //  proj phase : wbuf[2][192][72] bf16 (double-buffered weight chunks, K rows 0-95, V rows 96-191)
    //  post phase : kbuf[128][104] bf16 (0..26624) | accum_lds[64*96] f32 | count_lds[64] f32
    __shared__ __align__(16) unsigned char smem[55296];
    bf16_t* wb0 = (bf16_t*)smem;                 // [192][72]
    bf16_t* wb1 = wb0 + 192 * 72;
    bf16_t (*kbuf)[104] = (bf16_t (*)[104])smem;
    float* accum_lds = (float*)(smem + 26624);
    float* count_lds = (float*)(smem + 26624 + 24576);

    const int tid  = threadIdx.x;
    const int lane = tid & 31;
    const int wave = tid >> 5;
    const int col  = lane & 15;
    const int hi16 = lane >> 4;
    const int b    = blockIdx.y;
    const int s0   = blockIdx.x * 128 + wave * 16;

    const float* keyRow = key_in + ((size_t)b * S_ + (s0 + col)) * C_;

    for (int h = 0; h < H_; ++h) {
        const bf16_t* WkH = WkB + (size_t)(h * HD_) * C_;
        const bf16_t* WvH = WvB + (size_t)(h * HD_) * C_;

        // cooperative async stage of one 64-c chunk: 192 rows x 128B = 1536 x 16B
        auto stage = [&](bf16_t* dst, int cbase) {
            for (int j = 0; j < 6; ++j) {
                int idx = tid + j * 256;
                int row = idx >> 3;           // 0..191
                int seg = idx & 7;            // 16B segment
                const bf16_t* src = (row < 96 ? WkH + (size_t)row * C_
                                              : WvH + (size_t)(row - 96) * C_)
                                    + cbase + seg * 8;
                async_copy16(dst + row * 72 + seg * 8, src);
            }
        };

        stage(wb0, 0);

        // ---- K & V projections: [16 tok x 768] x [768 x 96] each ----
        v8f accK[6], accV[6];
        for (int nt = 0; nt < 6; ++nt)
            for (int r = 0; r < 8; ++r) { accK[nt][r] = 0.f; accV[nt][r] = 0.f; }

        for (int chunk = 0; chunk < 12; ++chunk) {
            bf16_t* wb = (chunk & 1) ? wb1 : wb0;
            bf16_t* wn = (chunk & 1) ? wb0 : wb1;
            wait_async0();          // own async writes done
            __syncthreads();        // everyone's writes done; readers of wn finished
            if (chunk + 1 < 12) stage(wn, (chunk + 1) * 64);
            for (int k2 = 0; k2 < 2; ++k2) {
                int kk = chunk * 2 + k2;
                v16bf a = make_a_from_f32(keyRow, kk * 32 + hi16 * 8);
                for (int nt = 0; nt < 6; ++nt) {
                    const bf16_t* pk = wb + (size_t)(nt * 16 + col) * 72 + k2 * 32 + hi16 * 8;
                    const bf16_t* pv = wb + (size_t)(96 + nt * 16 + col) * 72 + k2 * 32 + hi16 * 8;
                    accK[nt] = wmma_bf16(a, load_bf16_frag(pk, pk + 16), accK[nt]);
                    accV[nt] = wmma_bf16(a, load_bf16_frag(pv, pv + 16), accV[nt]);
                }
            }
        }
        __syncthreads();            // weight buffers dead; region becomes kbuf/accum

        // ---- reshape K D-tiles through LDS (bf16) for logits B-fragments ----
        for (int nt = 0; nt < 6; ++nt)
            for (int r = 0; r < 8; ++r) {
                int tok = hi16 * 8 + r;
                kbuf[wave * 16 + tok][nt * 16 + col] = (bf16_t)accK[nt][r];
            }
        // ---- zero per-head group accumulators (disjoint from kbuf) ----
        for (int i = tid; i < N_ * HD_; i += 256) accum_lds[i] = 0.f;
        if (tid < N_) count_lds[tid] = 0.f;
        // producer/consumer of kbuf are the same wave: drain DS, no barrier
        __asm__ volatile("s_wait_dscnt 0" ::: "memory");

        // ---- logits: [64 groups x 96] x [96 x 16 tok] ----
        v8f accL[4];
        for (int m = 0; m < 4; ++m)
            for (int r = 0; r < 8; ++r) accL[m][r] = 0.f;

        const bf16_t* qH = qB + (size_t)(b * H_ + h) * N_ * HD_;
        for (int kk = 0; kk < 3; ++kk) {
            const bf16_t* pb = &kbuf[wave * 16 + col][kk * 32 + hi16 * 8];
            v16bf bfrag = load_bf16_frag(pb, pb + 16);
            for (int m = 0; m < 4; ++m) {
                const bf16_t* pa = qH + (size_t)(m * 16 + col) * HD_ + kk * 32 + hi16 * 8;
                v16bf afrag = load_bf16_frag(pa, pa + 16);
                accL[m] = wmma_bf16(afrag, bfrag, accL[m]);
            }
        }

        // ---- argmax over 64 groups for token t = col ----
        float best = -3.0e38f;
        int   gbest = 0;
        for (int m = 0; m < 4; ++m)
            for (int r = 0; r < 8; ++r) {
                int g = m * 16 + hi16 * 8 + r;
                float v = accL[m][r];
                if (v > best) { best = v; gbest = g; }
            }
        float obest = __shfl_xor(best, 16);
        int   og    = __shfl_xor(gbest, 16);
        if (obest > best || (obest == best && og < gbest)) { best = obest; gbest = og; }

        int grp[8];
        for (int r = 0; r < 8; ++r) grp[r] = __shfl(gbest, hi16 * 8 + r);

        __syncthreads();            // zeros visible before scatter

        // ---- scatter V into LDS group accumulator (ds float atomics) ----
        for (int nt = 0; nt < 6; ++nt) {
            int d = nt * 16 + col;
            for (int r = 0; r < 8; ++r)
                unsafeAtomicAdd(&accum_lds[grp[r] * HD_ + d], accV[nt][r]);
        }
        if (col == 0) {             // lanes 0 and 16: tokens 0-7 / 8-15
            for (int r = 0; r < 8; ++r)
                unsafeAtomicAdd(&count_lds[grp[r]], 1.0f);
        }
        __syncthreads();

        // ---- flush LDS accumulator to global ----
        float* aG = accumG + (size_t)(b * H_ + h) * N_ * HD_;
        for (int i = tid; i < N_ * HD_; i += 256)
            unsafeAtomicAdd(&aG[i], accum_lds[i]);
        if (tid < N_)
            unsafeAtomicAdd(&countG[(b * H_ + h) * N_ + tid], count_lds[tid]);
        __syncthreads();            // region free before next head's stage
    }
}

// ---------------------------------------------------------------------------
// Finalize on WMMA: out[b] = (accum/(count+1)) @ Wo^T + bo, [64 x 768]x[768 x 768].
// Scale folded into A-fragment build; 96 = 3*32 so h = kk/3 within a chunk.
// Grid (2, B).
// ---------------------------------------------------------------------------
__global__ __launch_bounds__(256) void finalize_wmma(
    const float* __restrict__ accumG, const float* __restrict__ countG,
    const bf16_t* __restrict__ WoB, const float* __restrict__ bo,
    float* __restrict__ out)
{
    const int tid  = threadIdx.x;
    const int lane = tid & 31;
    const int wave = tid >> 5;
    const int col  = lane & 15;
    const int hi16 = lane >> 4;
    const int b    = blockIdx.y;

    // per-lane scales for the 4 M-tiles x 8 heads it contributes A rows for
    float invs[4][8];
    for (int mt = 0; mt < 4; ++mt)
        for (int h = 0; h < H_; ++h)
            invs[mt][h] = 1.0f / (countG[(b * H_ + h) * N_ + mt * 16 + col] + 1.0f);

    v8f acc[4][3];
    for (int mt = 0; mt < 4; ++mt)
        for (int j = 0; j < 3; ++j)
            for (int r = 0; r < 8; ++r) acc[mt][j][r] = 0.f;

    for (int kk = 0; kk < 24; ++kk) {
        int h  = kk / 3;
        int c0 = (kk % 3) * 32 + hi16 * 8;
        v16bf a[4];
        for (int mt = 0; mt < 4; ++mt) {
            const float* row = accumG + ((size_t)(b * H_ + h) * N_ + mt * 16 + col) * HD_;
            a[mt] = make_a_scaled(row, c0, invs[mt][h]);
        }
        for (int j = 0; j < 3; ++j) {
            int dg0 = (blockIdx.x * 24 + wave * 3 + j) * 16;
            const bf16_t* pb = WoB + (size_t)(dg0 + col) * C_ + kk * 32 + hi16 * 8;
            v16bf bf = load_bf16_frag(pb, pb + 16);
            for (int mt = 0; mt < 4; ++mt)
                acc[mt][j] = wmma_bf16(a[mt], bf, acc[mt][j]);
        }
    }
    for (int j = 0; j < 3; ++j) {
        int dg = (blockIdx.x * 24 + wave * 3 + j) * 16 + col;
        float bias = bo[dg];
        for (int mt = 0; mt < 4; ++mt)
            for (int r = 0; r < 8; ++r) {
                int n = mt * 16 + hi16 * 8 + r;
                out[(size_t)(b * N_ + n) * C_ + dg] = acc[mt][j][r] + bias;
            }
    }
}

// ---------------------------------------------------------------------------
extern "C" void kernel_launch(void* const* d_in, const int* in_sizes, int n_in,
                              void* d_out, int out_size, void* d_ws, size_t ws_size,
                              hipStream_t stream) {
    const float* query  = (const float*)d_in[0];
    const float* key_in = (const float*)d_in[1];
    const float* Wq     = (const float*)d_in[2];
    const float* Wk     = (const float*)d_in[3];
    const float* Wv     = (const float*)d_in[4];
    const float* Wo     = (const float*)d_in[5];
    const float* bo     = (const float*)d_in[6];

    char* ws = (char*)d_ws;
    // Workspace: accumF | countF | WkB | WvB | WqB | WoB | qB  (~9.0 MB)
    float*  accumG = (float*)(ws + 0);                 // 786432 f32
    float*  countG = (float*)(ws + 3145728);           // 8192 f32
    bf16_t* WkB    = (bf16_t*)(ws + 3178496);          // C*C bf16
    bf16_t* WvB    = (bf16_t*)(ws + 4358144);
    bf16_t* WqB    = (bf16_t*)(ws + 5537792);
    bf16_t* WoB    = (bf16_t*)(ws + 6717440);
    bf16_t* qB     = (bf16_t*)(ws + 7897088);          // B*H*N*HD bf16

    const int nAccum = B_ * H_ * N_ * HD_ + B_ * H_ * N_;   // 794624
    const int nW     = C_ * C_;                             // 589824

    prep_zero<<<(nAccum + 255) / 256, 256, 0, stream>>>(accumG, nAccum);
    prep_cvt4<<<(nW + 255) / 256, 256, 0, stream>>>(Wq, Wk, Wv, Wo, WqB, WkB, WvB, WoB, nW);

    qproj_wmma<<<dim3(2, B_), 256, 0, stream>>>(query, WqB, qB);

    assign_attn_main<<<dim3(S_ / 128, B_), 256, 0, stream>>>(
        key_in, WkB, WvB, qB, accumG, countG);

    finalize_wmma<<<dim3(2, B_), 256, 0, stream>>>(accumG, countG, WoB, bo, (float*)d_out);
}
